// TransformerAttention_36979668418858
// MI455X (gfx1250) — matrix operand order repaired and checked by reading, not compiled, not run on hardware
//
#include <hip/hip_runtime.h>

typedef __attribute__((ext_vector_type(16))) _Float16 v16h;
typedef __attribute__((ext_vector_type(8)))  _Float16 v8h;
typedef __attribute__((ext_vector_type(8)))  float    v8f;
typedef __attribute__((ext_vector_type(4)))  int      v4i;

#define HID 768
#define SEQ 2048
#define NH  12
#define HD  64

// --------------------------------------------------------------------------
// CDNA5 async global->LDS path (ASYNCcnt-tracked, ISA ch.10/15.18).
// Builtin signature (from hipcc diagnostic): first param is a pointer to a
// 4-int vector in the global address space; second is the LDS destination.
// --------------------------------------------------------------------------
#if defined(__has_builtin)
#if __has_builtin(__builtin_amdgcn_global_load_async_to_lds_b128) && \
    __has_builtin(__builtin_amdgcn_s_wait_asynccnt)
#define USE_ASYNC_LDS 1
#endif
#endif

#ifdef USE_ASYNC_LDS
typedef __attribute__((address_space(1))) v4i* g_v4i_ptr;
typedef __attribute__((address_space(3))) v4i* l_v4i_ptr;
__device__ __forceinline__ void async_b128(_Float16* lds, const _Float16* g) {
  __builtin_amdgcn_global_load_async_to_lds_b128(
      (g_v4i_ptr)(g), (l_v4i_ptr)(lds), 0, 0);
}
__device__ __forceinline__ void async_wait0() {
  __builtin_amdgcn_s_wait_asynccnt(0);
}
#else
__device__ __forceinline__ void async_b128(_Float16* lds, const _Float16* g) {
  *(v8h*)lds = *(const v8h*)g;
}
__device__ __forceinline__ void async_wait0() {}
#endif

// ---------------------------------------------------------------------------
// WMMA wrapper: D = A(16x32 f16) * B(32x16 f16) + C(16x16 f32)
// ---------------------------------------------------------------------------
__device__ __forceinline__ v8f wmma16(v16h a, v16h b, v8f c) {
  return __builtin_amdgcn_wmma_f32_16x16x32_f16(
      /*neg_a=*/false, a, /*neg_b=*/false, b,
      /*c_mod=*/(short)0, c, /*reuse_a=*/false, /*reuse_b=*/false);
}

// A-fragment (16x32, row-major LDS tile). Lane layout (ISA 7.12.2):
//   m = lane%16, half = lane/16; element e: K = (e/8)*16 + half*8 + (e%8)
__device__ __forceinline__ v16h fragA(const _Float16* tile, int stride,
                                      int mrow, int k0, int half) {
  const _Float16* p = tile + mrow * stride + k0 + half * 8;
  v8h lo = *(const v8h*)p;
  v8h hi = *(const v8h*)(p + 16);
  return __builtin_shufflevector(lo, hi, 0,1,2,3,4,5,6,7,8,9,10,11,12,13,14,15);
}

// B-fragment (32x16) from TRANSPOSED LDS tile Bt[n][k]:
//   n = lane%16, half = lane/16; element e: K = (e/8)*16 + half*8 + (e%8)
__device__ __forceinline__ v16h fragB(const _Float16* tileT, int stride,
                                      int ncol, int k0, int half) {
  const _Float16* p = tileT + ncol * stride + k0 + half * 8;
  v8h lo = *(const v8h*)p;
  v8h hi = *(const v8h*)(p + 16);
  return __builtin_shufflevector(lo, hi, 0,1,2,3,4,5,6,7,8,9,10,11,12,13,14,15);
}

// ---------------------------------------------------------------------------
// f32 -> f16 conversion
// ---------------------------------------------------------------------------
__global__ void f32_to_f16(const float* __restrict__ in,
                           _Float16* __restrict__ out, int n) {
  int i = blockIdx.x * blockDim.x + threadIdx.x;
  if (i < n) out[i] = (_Float16)in[i];
}

// ---------------------------------------------------------------------------
// GEMM: out[M x N] = A[M x K] * W[K x N] + bias
// Block tile 128(M) x 64(N), 256 threads = 8 waves arranged 4x2,
// each wave owns a 32x32 region = 2x2 WMMA tiles (4 wmma / k-step,
// each fragment reused twice).
// OUT_F32 == 0 : f16 output, permuted to [B, H, S, D]  (QKV projections)
// OUT_F32 == 1 : f32 output, row-major [M, N]          (final projection)
// ---------------------------------------------------------------------------
template <int OUT_F32>
__global__ __launch_bounds__(256)
void gemm128(const _Float16* __restrict__ A, const _Float16* __restrict__ W,
             const float* __restrict__ bias, void* __restrict__ outp,
             int M, int N, int Kdim) {
  constexpr int ASTR = 40, BSTR = 40;
  __shared__ __align__(16) _Float16 As[128 * ASTR];  // A tile row-major 128x32
  __shared__ __align__(16) _Float16 Bt[64 * BSTR];   // W tile transposed [n][k]

  const int tid  = threadIdx.x;
  const int lane = tid & 31;
  const int wv   = tid >> 5;
  const int half = lane >> 4, ln = lane & 15;
  const int wm = (wv >> 1) * 32;     // wave row offset in 128
  const int wn = (wv & 1) * 32;      // wave col offset in 64
  const int m0 = blockIdx.x * 128, n0 = blockIdx.y * 64;

  // loop-invariant copy indices
  const int arow = tid >> 1, aseg = tid & 1;       // A: 16 halfs per thread
  const int wk = tid >> 3,  wseg = tid & 7;        // W: 8 halfs per thread

  v8f acc[2][2] = {};
  const int nkb = Kdim / 32;

  for (int kb = 0; kb < nkb; ++kb) {
    __syncthreads();
    {  // A tile: 128 rows x 32 cols, two async 16B chunks per thread
      const _Float16* g = A + (size_t)(m0 + arow) * Kdim + kb * 32 + aseg * 16;
      _Float16* l = &As[arow * ASTR + aseg * 16];
      async_b128(l, g);
      async_b128(l + 8, g + 8);
    }
    {  // W tile: 32 k x 64 n, store transposed (scatter)
      const _Float16* g = W + (size_t)(kb * 32 + wk) * N + n0 + wseg * 8;
      v8h w = *(const v8h*)g;
#pragma unroll
      for (int j = 0; j < 8; ++j) Bt[(wseg * 8 + j) * BSTR + wk] = w[j];
    }
    if (kb + 1 < nkb) {  // prefetch next k-block
      __builtin_prefetch(A + (size_t)(m0 + arow) * Kdim + (kb + 1) * 32 + aseg * 16, 0, 0);
      __builtin_prefetch(W + (size_t)((kb + 1) * 32 + wk) * N + n0 + wseg * 8, 0, 0);
    }
    async_wait0();
    __syncthreads();

    v16h a0 = fragA(As, ASTR, wm + ln,      0, half);
    v16h a1 = fragA(As, ASTR, wm + 16 + ln, 0, half);
    v16h b0 = fragB(Bt, BSTR, wn + ln,      0, half);
    v16h b1 = fragB(Bt, BSTR, wn + 16 + ln, 0, half);
    acc[0][0] = wmma16(a0, b0, acc[0][0]);
    acc[0][1] = wmma16(a0, b1, acc[0][1]);
    acc[1][0] = wmma16(a1, b0, acc[1][0]);
    acc[1][1] = wmma16(a1, b1, acc[1][1]);
  }

  // C/D layout: VGPR r -> m = half*8 + r, n = lane%16
#pragma unroll
  for (int ti = 0; ti < 2; ++ti) {
#pragma unroll
    for (int tj = 0; tj < 2; ++tj) {
#pragma unroll
      for (int r = 0; r < 8; ++r) {
        int row = m0 + wm + ti * 16 + half * 8 + r;
        int c   = n0 + wn + tj * 16 + ln;
        float v = acc[ti][tj][r] + bias[c];
        if (OUT_F32) {
          ((float*)outp)[(size_t)row * N + c] = v;
        } else {
          int b = row >> 11, s = row & (SEQ - 1);
          int h = c >> 6, d = c & 63;
          ((_Float16*)outp)[((((size_t)b * NH + h) << 11) + s) * HD + d] =
              (_Float16)v;
        }
      }
    }
  }
}

// ---------------------------------------------------------------------------
// Flash attention: grid.x = S/64 (query blocks), grid.y = B*H.
// Q/K/V are [B,H,S,D] f16. Output Aout is [B,S,H*D] f16.
// ---------------------------------------------------------------------------
__global__ __launch_bounds__(256)
void attn64(const _Float16* __restrict__ Q, const _Float16* __restrict__ Kg,
            const _Float16* __restrict__ V, _Float16* __restrict__ Aout) {
  constexpr int QSTR = 72, KSTR = 72, VSTR = 72, PSTR = 72, SSTR = 65;
  __shared__ __align__(16) _Float16 Qb[64 * QSTR];  // Q tile [q][d]
  __shared__ __align__(16) _Float16 Kb[64 * KSTR];  // K tile [key][d] (= B^T)
  __shared__ __align__(16) _Float16 Vt[64 * VSTR];  // V tile transposed [d][key]
  __shared__ __align__(16) _Float16 Pt[64 * PSTR];  // P = exp(S-max) f16
  __shared__ __align__(16) float    Sf[64 * SSTR];  // raw scores f32
  __shared__ float m_st[64], l_st[64], sc_sh[64];

  const int tid  = threadIdx.x;
  const int lane = tid & 31;
  const int wv   = tid >> 5;
  const int half = lane >> 4, ln = lane & 15;
  const int t0 = wv * 2;
  const int tm = t0 >> 2, tn0 = t0 & 3, tn1 = (t0 + 1) & 3;

  const int q0 = blockIdx.x * 64;
  const int bh = blockIdx.y;
  const int bb = bh / NH, hh = bh % NH;
  const _Float16* Qp = Q  + (size_t)bh * SEQ * HD;
  const _Float16* Kp = Kg + (size_t)bh * SEQ * HD;
  const _Float16* Vp = V  + (size_t)bh * SEQ * HD;

  const int crow = tid >> 2, cseg = tid & 3;  // 16-half chunk copy indices

  {  // load Q block once (async 16B chunks)
    const _Float16* g = Qp + (size_t)(q0 + crow) * HD + cseg * 16;
    _Float16* l = &Qb[crow * QSTR + cseg * 16];
    async_b128(l, g);
    async_b128(l + 8, g + 8);
  }
  if (tid < 64) { m_st[tid] = -1e30f; l_st[tid] = 0.f; }
  async_wait0();

  v8f o0 = {}, o1 = {};

  for (int kb = 0; kb < SEQ / 64; ++kb) {
    __syncthreads();  // previous iteration fully consumed Kb/Vt/Pt
    {
      const _Float16* g = Kp + (size_t)(kb * 64 + crow) * HD + cseg * 16;
      _Float16* l = &Kb[crow * KSTR + cseg * 16];
      async_b128(l, g);
      async_b128(l + 8, g + 8);
      const _Float16* gv = Vp + (size_t)(kb * 64 + crow) * HD + cseg * 16;
      v8h a0 = *(const v8h*)gv;
      v8h a1 = *(const v8h*)(gv + 8);
#pragma unroll
      for (int j = 0; j < 8; ++j) Vt[(cseg * 16 + j) * VSTR + crow] = a0[j];
#pragma unroll
      for (int j = 0; j < 8; ++j) Vt[(cseg * 16 + 8 + j) * VSTR + crow] = a1[j];
    }
    if (kb + 1 < SEQ / 64) {  // prefetch next K/V block
      __builtin_prefetch(Kp + (size_t)((kb + 1) * 64 + crow) * HD + cseg * 16, 0, 0);
      __builtin_prefetch(Vp + (size_t)((kb + 1) * 64 + crow) * HD + cseg * 16, 0, 0);
    }
    async_wait0();
    __syncthreads();

    // S = Q * K^T   (head_dim 64 -> two k-steps of 32)
    v8f s0 = {}, s1 = {};
#pragma unroll
    for (int k0 = 0; k0 < 64; k0 += 32) {
      v16h a  = fragA(Qb, QSTR, tm * 16 + ln, k0, half);
      v16h b0 = fragB(Kb, KSTR, tn0 * 16 + ln, k0, half);
      v16h b1 = fragB(Kb, KSTR, tn1 * 16 + ln, k0, half);
      s0 = wmma16(a, b0, s0);
      s1 = wmma16(a, b1, s1);
    }
#pragma unroll
    for (int r = 0; r < 8; ++r) {
      int m = tm * 16 + half * 8 + r;
      Sf[m * SSTR + tn0 * 16 + ln] = s0[r];
      Sf[m * SSTR + tn1 * 16 + ln] = s1[r];
    }
    __syncthreads();

    // online softmax over this key block (one thread per query row)
    if (tid < 64) {
      int row = tid;
      float mo = m_st[row];
      float mx = mo;
#pragma unroll 8
      for (int j = 0; j < 64; ++j) mx = fmaxf(mx, Sf[row * SSTR + j]);
      float sc = __expf(mo - mx);
      float ls = 0.f;
#pragma unroll 8
      for (int j = 0; j < 64; ++j) {
        float p = __expf(Sf[row * SSTR + j] - mx);
        ls += p;
        Pt[row * PSTR + j] = (_Float16)p;
      }
      m_st[row]  = mx;
      l_st[row]  = l_st[row] * sc + ls;
      sc_sh[row] = sc;
    }
    __syncthreads();

    // rescale running output, then O += P * V
#pragma unroll
    for (int r = 0; r < 8; ++r) {
      float sc = sc_sh[tm * 16 + half * 8 + r];
      o0[r] *= sc;
      o1[r] *= sc;
    }
#pragma unroll
    for (int k0 = 0; k0 < 64; k0 += 32) {
      v16h a  = fragA(Pt, PSTR, tm * 16 + ln, k0, half);
      v16h b0 = fragB(Vt, VSTR, tn0 * 16 + ln, k0, half);
      v16h b1 = fragB(Vt, VSTR, tn1 * 16 + ln, k0, half);
      o0 = wmma16(a, b0, o0);
      o1 = wmma16(a, b1, o1);
    }
  }
  __syncthreads();

  // write normalized output to [B, S, H*D] f16
#pragma unroll
  for (int r = 0; r < 8; ++r) {
    int m = tm * 16 + half * 8 + r;
    float inv = 1.f / l_st[m];
    int s = q0 + m;
    size_t base = ((size_t)bb * SEQ + s) * HID + (size_t)hh * HD;
    Aout[base + tn0 * 16 + ln] = (_Float16)(o0[r] * inv);
    Aout[base + tn1 * 16 + ln] = (_Float16)(o1[r] * inv);
  }
}

// ---------------------------------------------------------------------------
// Launcher
// ---------------------------------------------------------------------------
extern "C" void kernel_launch(void* const* d_in, const int* in_sizes, int n_in,
                              void* d_out, int out_size, void* d_ws, size_t ws_size,
                              hipStream_t stream) {
  const float* x  = (const float*)d_in[0];
  const float* Wq = (const float*)d_in[1];
  const float* bq = (const float*)d_in[2];
  const float* Wk = (const float*)d_in[3];
  const float* bk = (const float*)d_in[4];
  const float* Wv = (const float*)d_in[5];
  const float* bv = (const float*)d_in[6];
  const float* Wo = (const float*)d_in[7];
  const float* bo = (const float*)d_in[8];

  const int M  = 2 * SEQ;          // 4096
  const int NX = M * HID;          // x / q / k / v / attn-out elements
  const int NW = HID * HID;        // weight elements

  _Float16* Xh  = (_Float16*)d_ws;
  _Float16* Wqh = Xh  + (size_t)NX;
  _Float16* Wkh = Wqh + (size_t)NW;
  _Float16* Wvh = Wkh + (size_t)NW;
  _Float16* Woh = Wvh + (size_t)NW;
  _Float16* Qh  = Woh + (size_t)NW;   // [B,H,S,D]
  _Float16* Kh  = Qh  + (size_t)NX;
  _Float16* Vh  = Kh  + (size_t)NX;
  _Float16* Ah  = Vh  + (size_t)NX;   // [B,S,H*D]

  f32_to_f16<<<(NX + 255) / 256, 256, 0, stream>>>(x,  Xh,  NX);
  f32_to_f16<<<(NW + 255) / 256, 256, 0, stream>>>(Wq, Wqh, NW);
  f32_to_f16<<<(NW + 255) / 256, 256, 0, stream>>>(Wk, Wkh, NW);
  f32_to_f16<<<(NW + 255) / 256, 256, 0, stream>>>(Wv, Wvh, NW);
  f32_to_f16<<<(NW + 255) / 256, 256, 0, stream>>>(Wo, Woh, NW);

  dim3 ggrid(M / 128, HID / 64);  // 32 x 12
  gemm128<0><<<ggrid, 256, 0, stream>>>(Xh, Wqh, bq, Qh, M, HID, HID);
  gemm128<0><<<ggrid, 256, 0, stream>>>(Xh, Wkh, bk, Kh, M, HID, HID);
  gemm128<0><<<ggrid, 256, 0, stream>>>(Xh, Wvh, bv, Vh, M, HID, HID);

  attn64<<<dim3(SEQ / 64, 2 * NH), 256, 0, stream>>>(Qh, Kh, Vh, Ah);

  gemm128<1><<<ggrid, 256, 0, stream>>>(Ah, Woh, bo, d_out, M, HID, HID);
}
